// MultiHeadAttention_57793079935624
// MI455X (gfx1250) — matrix operand order repaired and checked by reading, not compiled, not run on hardware
//
#include <hip/hip_runtime.h>

#define B_ 2
#define S_ 2048
#define D_ 1024
#define H_ 16
#define DH_ 64

typedef __bf16 bf16;
typedef __attribute__((ext_vector_type(16))) __bf16 v16bf;
typedef __attribute__((ext_vector_type(8)))  __bf16 v8bf;
typedef __attribute__((ext_vector_type(8)))  float  v8f;

union FragB16 { v16bf v; v8bf h[2]; };

static __device__ inline v8f vzero8() {
  v8f z;
#pragma unroll
  for (int i = 0; i < 8; ++i) z[i] = 0.0f;
  return z;
}

static __device__ inline v8f wmma_bf16(const FragB16& a, const FragB16& b, v8f c) {
  return __builtin_amdgcn_wmma_f32_16x16x32_bf16(false, a.v, false, b.v,
                                                 (short)0, c, false, false);
}

// 16-byte async global->LDS copy (ASYNCcnt-tracked).  LDS byte address =
// low 32 bits of the flat pointer (flat LDS aperture: addr[31:0] -> LDS).
static __device__ inline void async_g2l_b128(void* lds_dst, const bf16* gsrc) {
  unsigned lds_addr = (unsigned)(size_t)lds_dst;
  unsigned long long gaddr = (unsigned long long)(size_t)gsrc;
  asm volatile("global_load_async_to_lds_b128 %0, %1, off"
               :: "v"(lds_addr), "v"(gaddr) : "memory");
}
static __device__ inline void wait_async0() {
  asm volatile("s_wait_asynccnt 0x0" ::: "memory");
}

// ---------------------------------------------------------------- conversions

__global__ void mha_cvt_bf16_kernel(const float* __restrict__ in,
                                    bf16* __restrict__ out, int n) {
  int i = blockIdx.x * blockDim.x + threadIdx.x;
  if (i < n) out[i] = (bf16)in[i];
}

// out[bz][c][r] = (bf16) in[bz][r][c]
__global__ void mha_tcvt_bf16_kernel(const float* __restrict__ in,
                                     bf16* __restrict__ out, int R, int C) {
  int bz = blockIdx.y;
  int i = blockIdx.x * blockDim.x + threadIdx.x;
  if (i >= R * C) return;
  int r = i / C, c = i % C;
  out[((size_t)bz * C + c) * R + r] = (bf16)in[((size_t)bz * R + r) * C + c];
}

// ------------------------------------------------------------- Q/K projection
// Out[bh][s][e] = sum_d X[b][s][d] * W[h][d][e];  Wt is pre-transposed [H][DH][D].
// B tile (64 rows x 32 k) staged via double-buffered async-LDS copies.
__global__ __launch_bounds__(256) void mha_proj_kernel(
    const bf16* __restrict__ X,    // [B][S][D]
    const bf16* __restrict__ Wt,   // [H][DH][D]
    bf16* __restrict__ Out,        // [B*H][S][DH]
    bf16* __restrict__ OutT)       // [B*H][DH][S] or nullptr
{
  __shared__ bf16 btile[2][64 * 32];           // 2 x 4 KB

  const int tid  = threadIdx.x;
  const int w    = tid >> 5;
  const int lane = tid & 31;
  const int l15  = lane & 15;
  const int hi   = lane >> 4;
  const int bh   = blockIdx.y;
  const int b    = bh / H_;
  const int h    = bh % H_;
  const int row0 = blockIdx.x * 128 + w * 16;

  const bf16* A  = X  + (size_t)b * S_ * D_;
  const bf16* Bt = Wt + (size_t)h * DH_ * D_;

  // this thread's slice of the 64x32 B tile: one b128 per chunk
  const int te = tid >> 2;                     // tile row (e)
  const int tc = (tid & 3) * 8;                // tile col (k element)
  const bf16* bsrc = Bt + (size_t)te * D_ + tc;

  v8f acc[4];
#pragma unroll
  for (int i = 0; i < 4; ++i) acc[i] = vzero8();

  const bf16* arow = A + (size_t)(row0 + l15) * D_;

  const int NCH = D_ / 32;
  // preamble: stage chunk 0, load A fragment 0
  async_g2l_b128(&btile[0][te * 32 + tc], bsrc);
  FragB16 a_cur;
  a_cur.h[0] = *(const v8bf*)(arow + hi * 8);
  a_cur.h[1] = *(const v8bf*)(arow + hi * 8 + 16);

  for (int i = 0; i < NCH; ++i) {
    wait_async0();                             // this wave's copy of chunk i done
    __syncthreads();                           // all copies visible; prev reads done

    FragB16 a_nxt = a_cur;
    if (i + 1 < NCH) {
      const int kk = (i + 1) * 32;
      async_g2l_b128(&btile[(i + 1) & 1][te * 32 + tc], bsrc + kk);
      const bf16* ap = arow + kk + hi * 8;
      a_nxt.h[0] = *(const v8bf*)(ap);
      a_nxt.h[1] = *(const v8bf*)(ap + 16);
    }

    const bf16* lb = &btile[i & 1][0];
    FragB16 bm[4];
#pragma unroll
    for (int nt = 0; nt < 4; ++nt) {           // B 32x16: lane=col, K = hi*8..+7, 16+hi*8..+7
      const bf16* bp = lb + (nt * 16 + l15) * 32 + hi * 8;
      bm[nt].h[0] = *(const v8bf*)(bp);
      bm[nt].h[1] = *(const v8bf*)(bp + 16);
    }
#pragma unroll
    for (int nt = 0; nt < 4; ++nt)
      acc[nt] = wmma_bf16(a_cur, bm[nt], acc[nt]);

    a_cur = a_nxt;
  }

  bf16* o = Out + (size_t)bh * S_ * DH_;
#pragma unroll
  for (int nt = 0; nt < 4; ++nt) {
#pragma unroll
    for (int r = 0; r < 8; ++r) {
      const int m   = r + hi * 8;              // C layout: M = r + hi*8, N = l15
      const int col = nt * 16 + l15;
      const bf16 v  = (bf16)acc[nt][r];
      o[(size_t)(row0 + m) * DH_ + col] = v;
      if (OutT)
        OutT[(size_t)bh * DH_ * S_ + (size_t)col * S_ + (row0 + m)] = v;
    }
  }
}

// --------------------------------------------------------- flash attention
// scores = (Q K^T) * scale, causal; O = softmax(scores) @ Q   (v == q bug kept)
__global__ __launch_bounds__(256) void mha_flash_kernel(
    const bf16* __restrict__ Q,    // [BH][S][DH]
    const bf16* __restrict__ Qt,   // [BH][DH][S]
    const bf16* __restrict__ K,    // [BH][S][DH]
    bf16* __restrict__ AttnB)      // [B][S][H*DH]
{
  __shared__ bf16 pbuf[8 * 16 * 32];           // per-wave 16x32 P tile (8 KB)

  const int w    = threadIdx.x >> 5;
  const int lane = threadIdx.x & 31;
  const int l15  = lane & 15;
  const int hi   = lane >> 4;
  const int bh   = blockIdx.y;
  const int b    = bh / H_;
  const int h    = bh % H_;
  const int q0   = blockIdx.x * 128 + w * 16;

  const bf16* Qb  = Q  + (size_t)bh * S_ * DH_;
  const bf16* Qtb = Qt + (size_t)bh * DH_ * S_;
  const bf16* Kb  = K  + (size_t)bh * S_ * DH_;
  bf16* pw = pbuf + w * (16 * 32);

  // Preload this wave's 16xDH Q strip as two A-fragments (e-chunks 0,32).
  FragB16 aq[2];
  {
    const bf16* qp = Qb + (size_t)(q0 + l15) * DH_;
#pragma unroll
    for (int ec = 0; ec < 2; ++ec) {
      const bf16* p = qp + ec * 32 + hi * 8;
      aq[ec].h[0] = *(const v8bf*)(p);
      aq[ec].h[1] = *(const v8bf*)(p + 16);
    }
  }

  v8f o[4];
#pragma unroll
  for (int i = 0; i < 4; ++i) o[i] = vzero8();
  float mrun[8], lrun[8];
#pragma unroll
  for (int r = 0; r < 8; ++r) { mrun[r] = -__builtin_inff(); lrun[r] = 0.0f; }

  const float scale = 0.03125f;                // 1024^-0.5
  const int   tend  = q0 + 15;                 // causal: keys <= row

  for (int t0 = 0; t0 <= tend; t0 += 32) {
    // prefetch next key chunk (speculative; in-allocation)
    __builtin_prefetch(Kb + (size_t)(t0 + 32 + l15) * DH_, 0, 0);

    // ---- load all 4 K B-fragments, then 4 score WMMAs back-to-back
    FragB16 bk[2][2];                          // [tile a/b][e-chunk]
#pragma unroll
    for (int ec = 0; ec < 2; ++ec) {
      const bf16* pa = Kb + (size_t)(t0 + l15) * DH_ + ec * 32 + hi * 8;
      bk[0][ec].h[0] = *(const v8bf*)(pa);
      bk[0][ec].h[1] = *(const v8bf*)(pa + 16);
      const bf16* pb = Kb + (size_t)(t0 + 16 + l15) * DH_ + ec * 32 + hi * 8;
      bk[1][ec].h[0] = *(const v8bf*)(pb);
      bk[1][ec].h[1] = *(const v8bf*)(pb + 16);
    }
    v8f sa = vzero8(), sb = vzero8();
    sa = wmma_bf16(aq[0], bk[0][0], sa);
    sb = wmma_bf16(aq[0], bk[1][0], sb);
    sa = wmma_bf16(aq[1], bk[0][1], sa);
    sb = wmma_bf16(aq[1], bk[1][1], sb);

    // ---- causal mask + online softmax (row m lives in lanes with matching hi)
    float alpha[8];
#pragma unroll
    for (int r = 0; r < 8; ++r) {
      const int srow = q0 + r + hi * 8;
      float va = (t0 + l15      <= srow) ? sa[r] * scale : -__builtin_inff();
      float vb = (t0 + 16 + l15 <= srow) ? sb[r] * scale : -__builtin_inff();
      float rm = fmaxf(va, vb);
      rm = fmaxf(rm, __shfl_xor(rm, 1, 32));
      rm = fmaxf(rm, __shfl_xor(rm, 2, 32));
      rm = fmaxf(rm, __shfl_xor(rm, 4, 32));
      rm = fmaxf(rm, __shfl_xor(rm, 8, 32));
      const float mnew = fmaxf(mrun[r], rm);
      const float al   = __expf(mrun[r] - mnew);
      const float pa   = __expf(va - mnew);
      const float pb   = __expf(vb - mnew);
      float rs = pa + pb;
      rs += __shfl_xor(rs, 1, 32);
      rs += __shfl_xor(rs, 2, 32);
      rs += __shfl_xor(rs, 4, 32);
      rs += __shfl_xor(rs, 8, 32);
      lrun[r] = lrun[r] * al + rs;
      mrun[r] = mnew;
      alpha[r] = al;
      // stash P in row-major 16x32 (C-layout -> LDS)
      pw[(r + hi * 8) * 32 + l15]      = (bf16)pa;
      pw[(r + hi * 8) * 32 + 16 + l15] = (bf16)pb;
    }
#pragma unroll
    for (int nt = 0; nt < 4; ++nt)
#pragma unroll
      for (int r = 0; r < 8; ++r) o[nt][r] *= alpha[r];

    // cross-lane LDS round trip: wait split DS counter, reload in A-layout
    asm volatile("s_wait_dscnt 0" ::: "memory");
    FragB16 pf;
    {
      const bf16* p = pw + l15 * 32 + hi * 8;
      pf.h[0] = *(const v8bf*)(p);
      pf.h[1] = *(const v8bf*)(p + 16);
    }
    asm volatile("s_wait_dscnt 0" ::: "memory");

    // ---- O += P @ Q[t0:t0+32, :]  (B-fragments from Qt are contiguous rows)
    FragB16 bq[4];
#pragma unroll
    for (int nt = 0; nt < 4; ++nt) {
      const bf16* p = Qtb + (size_t)(nt * 16 + l15) * S_ + t0 + hi * 8;
      bq[nt].h[0] = *(const v8bf*)(p);
      bq[nt].h[1] = *(const v8bf*)(p + 16);
    }
#pragma unroll
    for (int nt = 0; nt < 4; ++nt)
      o[nt] = wmma_bf16(pf, bq[nt], o[nt]);
    asm volatile("" ::: "memory");
  }

  // ---- normalize and write concat-head layout [B][S][H*DH]
  bf16* ob = AttnB + (size_t)b * S_ * D_ + (size_t)h * DH_;
#pragma unroll
  for (int r = 0; r < 8; ++r) {
    const float inv = 1.0f / lrun[r];
    const int   m   = r + hi * 8;
#pragma unroll
    for (int nt = 0; nt < 4; ++nt)
      ob[(size_t)(q0 + m) * D_ + nt * 16 + l15] = (bf16)(o[nt][r] * inv);
  }
}

// ------------------------------------------------------------ output proj
// Out[s][n] = sum_k Attn[s][k] * Wo[k][n] + bo[n];  WoT is [n][k].
__global__ __launch_bounds__(256) void mha_outproj_kernel(
    const bf16* __restrict__ A,     // [B*S][D]
    const bf16* __restrict__ WoT,   // [D][D]
    const float* __restrict__ bias, // [D]
    float* __restrict__ Out)        // [B*S][D]
{
  __shared__ bf16 btile[2][64 * 32];

  const int tid  = threadIdx.x;
  const int w    = tid >> 5;
  const int lane = tid & 31;
  const int l15  = lane & 15;
  const int hi   = lane >> 4;
  const int r0   = blockIdx.x * 128 + w * 16;
  const int n0   = blockIdx.y * 64;

  const int te = tid >> 2;
  const int tc = (tid & 3) * 8;
  const bf16* bsrc = WoT + (size_t)(n0 + te) * D_ + tc;

  v8f acc[4];
#pragma unroll
  for (int i = 0; i < 4; ++i) acc[i] = vzero8();

  const bf16* arow = A + (size_t)(r0 + l15) * D_;

  const int NCH = D_ / 32;
  async_g2l_b128(&btile[0][te * 32 + tc], bsrc);
  FragB16 a_cur;
  a_cur.h[0] = *(const v8bf*)(arow + hi * 8);
  a_cur.h[1] = *(const v8bf*)(arow + hi * 8 + 16);

  for (int i = 0; i < NCH; ++i) {
    wait_async0();
    __syncthreads();

    FragB16 a_nxt = a_cur;
    if (i + 1 < NCH) {
      const int kk = (i + 1) * 32;
      async_g2l_b128(&btile[(i + 1) & 1][te * 32 + tc], bsrc + kk);
      const bf16* ap = arow + kk + hi * 8;
      a_nxt.h[0] = *(const v8bf*)(ap);
      a_nxt.h[1] = *(const v8bf*)(ap + 16);
    }

    const bf16* lb = &btile[i & 1][0];
    FragB16 bm[4];
#pragma unroll
    for (int nt = 0; nt < 4; ++nt) {
      const bf16* bp = lb + (nt * 16 + l15) * 32 + hi * 8;
      bm[nt].h[0] = *(const v8bf*)(bp);
      bm[nt].h[1] = *(const v8bf*)(bp + 16);
    }
#pragma unroll
    for (int nt = 0; nt < 4; ++nt)
      acc[nt] = wmma_bf16(a_cur, bm[nt], acc[nt]);

    a_cur = a_nxt;
  }

#pragma unroll
  for (int nt = 0; nt < 4; ++nt) {
#pragma unroll
    for (int r = 0; r < 8; ++r) {
      const int row = r0 + r + hi * 8;
      const int col = n0 + nt * 16 + l15;
      Out[(size_t)row * D_ + col] = acc[nt][r] + bias[col];
    }
  }
}

// ----------------------------------------------------------------- launcher

extern "C" void kernel_launch(void* const* d_in, const int* in_sizes, int n_in,
                              void* d_out, int out_size, void* d_ws, size_t ws_size,
                              hipStream_t stream) {
  (void)in_sizes; (void)n_in; (void)out_size; (void)ws_size;
  const float* x  = (const float*)d_in[0];
  /* d_in[1] = mask: causal handled analytically */
  const float* Wq = (const float*)d_in[2];
  const float* Wk = (const float*)d_in[3];
  const float* Wo = (const float*)d_in[4];
  const float* bo = (const float*)d_in[5];
  float* out = (float*)d_out;

  char* ws = (char*)d_ws;
  size_t off = 0;
  auto carve = [&](size_t bytes) -> char* {
    char* p = ws + off;
    off += (bytes + 255) & ~(size_t)255;
    return p;
  };
  bf16* xb  = (bf16*)carve((size_t)B_ * S_ * D_ * 2);
  bf16* WqT = (bf16*)carve((size_t)H_ * DH_ * D_ * 2);
  bf16* WkT = (bf16*)carve((size_t)H_ * DH_ * D_ * 2);
  bf16* WoT = (bf16*)carve((size_t)D_ * D_ * 2);
  bf16* Qb  = (bf16*)carve((size_t)B_ * H_ * S_ * DH_ * 2);
  bf16* QtB = (bf16*)carve((size_t)B_ * H_ * DH_ * S_ * 2);
  bf16* Kb  = (bf16*)carve((size_t)B_ * H_ * S_ * DH_ * 2);
  bf16* AtB = (bf16*)carve((size_t)B_ * S_ * D_ * 2);

  {
    int n = B_ * S_ * D_;
    mha_cvt_bf16_kernel<<<(n + 255) / 256, 256, 0, stream>>>(x, xb, n);
  }
  {
    dim3 g((D_ * DH_ + 255) / 256, H_);
    mha_tcvt_bf16_kernel<<<g, 256, 0, stream>>>(Wq, WqT, D_, DH_);
    mha_tcvt_bf16_kernel<<<g, 256, 0, stream>>>(Wk, WkT, D_, DH_);
  }
  {
    dim3 g((D_ * D_ + 255) / 256, 1);
    mha_tcvt_bf16_kernel<<<g, 256, 0, stream>>>(Wo, WoT, D_, D_);
  }

  dim3 gp(S_ / 128, B_ * H_);
  mha_proj_kernel<<<gp, 256, 0, stream>>>(xb, WqT, Qb, QtB);
  mha_proj_kernel<<<gp, 256, 0, stream>>>(xb, WkT, Kb, nullptr);
  mha_flash_kernel<<<gp, 256, 0, stream>>>(Qb, QtB, Kb, AtB);

  dim3 go((B_ * S_) / 128, D_ / 64);
  mha_outproj_kernel<<<go, 256, 0, stream>>>(AtB, WoT, bo, out);
}
